// TDNetDenoiser_5677946765710
// MI455X (gfx1250) — matrix operand order repaired and checked by reading, not compiled, not run on hardware
//
#include <hip/hip_runtime.h>
#include <cmath>
#include <cstdint>
#include <cstddef>

// ---------------------------------------------------------------------------
// Problem constants (match the JAX reference)
// ---------------------------------------------------------------------------
#define B_    8
#define N_    500
#define KNN_  20
#define D_    256
#define H_    8
#define L_    6
#define DK_   32
#define BN_   (B_ * N_)          // 4000 tokens
#define BNK_  (B_ * N_ * KNN_)   // 80000 edges
#define NPAD_ 512                // padded attention K-dim

typedef __attribute__((ext_vector_type(16))) _Float16 v16h;
typedef __attribute__((ext_vector_type(8)))  _Float16 v8h;
typedef __attribute__((ext_vector_type(8)))  float    v8f;

enum { EPI_NONE = 0, EPI_BIAS = 1, EPI_BIAS_RELU = 2, EPI_BN_LEAKY = 3 };

__device__ __forceinline__ v16h cat8(v8h a, v8h b)
{
    return __builtin_shufflevector(a, b, 0, 1, 2, 3, 4, 5, 6, 7,
                                         8, 9, 10, 11, 12, 13, 14, 15);
}

// ---------------------------------------------------------------------------
// Quad-tile WMMA core: computes the 32x32 output block (m0..m0+31, n0..n0+31)
// as four v_wmma_f32_16x16x32_f16 per K-step, sharing two A fragments and two
// B fragments (8x global_load_b128 -> 4 WMMA, 16 flop/byte/lane).
//
// Operand layout contract (all strides in f16 elements, multiples of 8):
//   A : row-major, K-contiguous, K zero-padded to mult. of 32. (m,k) = A[m*saM+k]
//   Bt: n-major,  K-contiguous, K zero-padded to mult. of 32. (k,n) = Bt[n*sbN+k]
//
// gfx1250 wave32 fragment layouts (CDNA5 ISA 7.12.2): each lane's A fragment is
// two contiguous 8-half chunks at k0+8*hi and k0+16+8*hi; each lane's B
// fragment is two contiguous 8-half chunks at k0+8*hi and k0+16+8*hi. All loads are aligned
// b128, no per-element predication. Row/col indices are clamped (duplicated
// results discarded at the guarded stores).
// ---------------------------------------------------------------------------
__device__ __forceinline__ void wmma_quad32x32(
    const _Float16* __restrict__ A, long saM,
    const _Float16* __restrict__ Bt, long sbN,
    int m0, int n0, int M, int N, int Kdim,
    v8f& acc00, v8f& acc01, v8f& acc10, v8f& acc11)
{
    const int lane = threadIdx.x & 31;
    const int hi   = lane >> 4;
    const int l15  = lane & 15;
    int row0 = m0 + l15;        if (row0 > M - 1) row0 = M - 1;
    int row1 = m0 + 16 + l15;   if (row1 > M - 1) row1 = M - 1;
    int col0 = n0 + l15;        if (col0 > N - 1) col0 = N - 1;
    int col1 = n0 + 16 + l15;   if (col1 > N - 1) col1 = N - 1;

    const _Float16* ap0 = A  + (long)row0 * saM + (hi << 3);
    const _Float16* ap1 = A  + (long)row1 * saM + (hi << 3);
    const _Float16* bp0 = Bt + (long)col0 * sbN + (hi << 3);
    const _Float16* bp1 = Bt + (long)col1 * sbN + (hi << 3);

    for (int k0 = 0; k0 < Kdim; k0 += 32) {
        const v16h af0 = cat8(*(const v8h*)(ap0 + k0), *(const v8h*)(ap0 + k0 + 16));
        const v16h af1 = cat8(*(const v8h*)(ap1 + k0), *(const v8h*)(ap1 + k0 + 16));
        const v16h bf0 = cat8(*(const v8h*)(bp0 + k0), *(const v8h*)(bp0 + k0 + 16));
        const v16h bf1 = cat8(*(const v8h*)(bp1 + k0), *(const v8h*)(bp1 + k0 + 16));
        acc00 = __builtin_amdgcn_wmma_f32_16x16x32_f16(false, af0, false, bf0,
                                                       (short)0, acc00, false, false);
        acc01 = __builtin_amdgcn_wmma_f32_16x16x32_f16(false, af0, false, bf1,
                                                       (short)0, acc01, false, false);
        acc10 = __builtin_amdgcn_wmma_f32_16x16x32_f16(false, af1, false, bf0,
                                                       (short)0, acc10, false, false);
        acc11 = __builtin_amdgcn_wmma_f32_16x16x32_f16(false, af1, false, bf1,
                                                       (short)0, acc11, false, false);
    }
}

// Guarded epilogue store for one 16x16 accumulator (C/D layout: VGPR i ->
// row m0+i+8*hi, col n0+(lane&15)).
__device__ __forceinline__ void store_tile(
    const v8f& acc, int m0, int n0, int M, int N, long ldc,
    float* __restrict__ Cf, _Float16* __restrict__ Ch,
    const float* __restrict__ bias, const float* __restrict__ resid, int mode,
    const float* __restrict__ bng, const float* __restrict__ bnbe,
    const float* __restrict__ bnm, const float* __restrict__ bnv)
{
    const int lane = threadIdx.x & 31;
    const int hi   = lane >> 4, l15 = lane & 15;
    const int col  = n0 + l15;
    if (col >= N) return;

    float badd = 0.f, scl = 1.f, shf = 0.f;
    if (mode != EPI_NONE && bias) badd = bias[col];
    if (mode == EPI_BN_LEAKY) {
        const float inv = rsqrtf(bnv[col] + 1e-5f);
        scl = bng[col] * inv;
        shf = bnbe[col] - bnm[col] * scl;
    }
#pragma unroll
    for (int i = 0; i < 8; ++i) {
        const int row = m0 + i + (hi << 3);
        if (row >= M) continue;
        float v = acc[i] + badd;
        if (mode == EPI_BN_LEAKY)       { v = v * scl + shf; v = (v > 0.f) ? v : 0.2f * v; }
        else if (mode == EPI_BIAS_RELU) { v = (v > 0.f) ? v : 0.f; }
        if (resid) v += resid[(long)row * ldc + col];
        if (Cf) Cf[(long)row * ldc + col] = v;
        if (Ch) Ch[(long)row * ldc + col] = (_Float16)v;
    }
}

// ---------------------------------------------------------------------------
// Generic WMMA GEMM: one 32x32 output block per wave, 8 waves / 256-thr block.
// ---------------------------------------------------------------------------
__global__ void gemm_f16_kernel(
    const _Float16* __restrict__ A, long saM,
    const _Float16* __restrict__ Bt, long sbN,
    float* __restrict__ Cf, _Float16* __restrict__ Ch, long ldc,
    const float* __restrict__ bias, const float* __restrict__ resid,
    int M, int N, int Kdim, int mode,
    const float* __restrict__ bng, const float* __restrict__ bnbe,
    const float* __restrict__ bnm, const float* __restrict__ bnv)
{
    const int tilesN = (N + 31) >> 5;
    const int tilesM = (M + 31) >> 5;
    const long wave  = (long)blockIdx.x * (blockDim.x >> 5) + (threadIdx.x >> 5);
    if (wave >= (long)tilesM * tilesN) return;
    const int tm = (int)(wave / tilesN);
    const int tn = (int)(wave % tilesN);
    const int m0 = tm << 5, n0 = tn << 5;

    v8f a00 = {}, a01 = {}, a10 = {}, a11 = {};
    wmma_quad32x32(A, saM, Bt, sbN, m0, n0, M, N, Kdim, a00, a01, a10, a11);

    store_tile(a00, m0,      n0,      M, N, ldc, Cf, Ch, bias, resid, mode, bng, bnbe, bnm, bnv);
    store_tile(a01, m0,      n0 + 16, M, N, ldc, Cf, Ch, bias, resid, mode, bng, bnbe, bnm, bnv);
    store_tile(a10, m0 + 16, n0,      M, N, ldc, Cf, Ch, bias, resid, mode, bng, bnbe, bnm, bnv);
    store_tile(a11, m0 + 16, n0 + 16, M, N, ldc, Cf, Ch, bias, resid, mode, bng, bnbe, bnm, bnv);
}

// ---------------------------------------------------------------------------
// Attention scores: S[b,h,nq,nk] = scale*(Q.K^T) + rel[h, rel_idx[b,nq,nk]]
// A = Q head slice (row-major, K-contig, saM=D). B = K head slice: element
// (d, nk) lives at k16[nk*D + d] -> exactly the Bt contract with sbN=D.
// ---------------------------------------------------------------------------
__global__ void attn_scores_kernel(
    const _Float16* __restrict__ q16, const _Float16* __restrict__ k16,
    const int* __restrict__ ridx, const float* __restrict__ rel,
    float* __restrict__ scores, float scale)
{
    const int bh = blockIdx.y;
    const int b = bh / H_, h = bh % H_;
    const int tilesN = (N_ + 31) >> 5;   // 16
    const int tilesM = (N_ + 31) >> 5;   // 16
    const long wave = (long)blockIdx.x * (blockDim.x >> 5) + (threadIdx.x >> 5);
    if (wave >= (long)tilesM * tilesN) return;
    const int tm = (int)(wave / tilesN), tn = (int)(wave % tilesN);
    const int m0 = tm << 5, n0 = tn << 5;

    const _Float16* A  = q16 + (long)b * N_ * D_ + h * DK_;
    const _Float16* Bt = k16 + (long)b * N_ * D_ + h * DK_;
    v8f a00 = {}, a01 = {}, a10 = {}, a11 = {};
    wmma_quad32x32(A, D_, Bt, D_, m0, n0, N_, N_, DK_, a00, a01, a10, a11);

    const int lane = threadIdx.x & 31;
    const int hi = lane >> 4, l15 = lane & 15;
#pragma unroll
    for (int qd = 0; qd < 4; ++qd) {
        const int mo = m0 + 16 * (qd >> 1);
        const int col = n0 + 16 * (qd & 1) + l15;
        if (col >= N_) continue;
        const v8f& acc = (qd == 0) ? a00 : (qd == 1) ? a01 : (qd == 2) ? a10 : a11;
#pragma unroll
        for (int i = 0; i < 8; ++i) {
            const int row = mo + i + (hi << 3);
            if (row >= N_) continue;
            const int q = ridx[((long)b * N_ + row) * N_ + col];
            scores[((long)bh * N_ + row) * N_ + col] = acc[i] * scale + rel[h * 64 + q];
        }
    }
}

// ---------------------------------------------------------------------------
// Attention context: ctx[b,n,h,:] = sum_nk P[b,h,n,nk] * V[b,nk,h,:]
// A = probs [N, NPAD] (zero-padded cols). B = vt [B*H, DK, NPAD] (K-contig,
// sbN=NPAD, pad nk>=500 zeroed). Kdim = NPAD exactly (pad contributes 0).
// One 32x32 block covers all DK=32 columns and 32 query rows.
// ---------------------------------------------------------------------------
__global__ void attn_ctx_kernel(
    const _Float16* __restrict__ probs, const _Float16* __restrict__ vt,
    _Float16* __restrict__ ctx16)
{
    const int bh = blockIdx.y;
    const int b = bh / H_, h = bh % H_;
    const int tilesM = (N_ + 31) >> 5;   // 16
    const long wave = (long)blockIdx.x * (blockDim.x >> 5) + (threadIdx.x >> 5);
    if (wave >= tilesM) return;
    const int m0 = (int)wave << 5;

    const _Float16* A  = probs + (long)bh * N_ * NPAD_;
    const _Float16* Bt = vt + (long)bh * DK_ * NPAD_;
    v8f a00 = {}, a01 = {}, a10 = {}, a11 = {};
    wmma_quad32x32(A, NPAD_, Bt, NPAD_, m0, 0, N_, DK_, NPAD_, a00, a01, a10, a11);

    const int lane = threadIdx.x & 31;
    const int hi = lane >> 4, l15 = lane & 15;
#pragma unroll
    for (int qd = 0; qd < 4; ++qd) {
        const int mo = m0 + 16 * (qd >> 1);
        const int col = 16 * (qd & 1) + l15;     // < DK_ always
        const v8f& acc = (qd == 0) ? a00 : (qd == 1) ? a01 : (qd == 2) ? a10 : a11;
#pragma unroll
        for (int i = 0; i < 8; ++i) {
            const int row = mo + i + (hi << 3);
            if (row >= N_) continue;
            ctx16[((long)b * N_ + row) * D_ + h * DK_ + col] = (_Float16)acc[i];
        }
    }
}

// ---------------------------------------------------------------------------
// KNN: one 256-thread block per (b,n); squared distances in LDS, K rounds of
// argmin with lowest-index tie-break (matches lax.top_k on -d2, self incl.)
// ---------------------------------------------------------------------------
__global__ void knn_kernel(const float* __restrict__ x, int* __restrict__ idxout,
                           int Cc)
{
    const int bn = blockIdx.x;           // b*N + n
    const int t  = threadIdx.x;
    __shared__ float ctr[64];
    __shared__ float d2[512];
    __shared__ float rv[256];
    __shared__ int   ri[256];

    const float* xb = x + (long)(bn / N_) * N_ * Cc;
    const int n = bn % N_;
    if (t < Cc) ctr[t] = xb[(long)n * Cc + t];
    __syncthreads();

    for (int j = t; j < 512; j += 256) {
        float s = 3.4e38f;
        if (j < N_) {
            s = 0.f;
            for (int c = 0; c < Cc; ++c) {
                const float dd = xb[(long)j * Cc + c] - ctr[c];
                s += dd * dd;
            }
        }
        d2[j] = s;
    }
    __syncthreads();

    for (int kk = 0; kk < KNN_; ++kk) {
        float v0 = d2[t], v1 = d2[t + 256];
        float bv = v0; int bi = t;
        if (v1 < bv) { bv = v1; bi = t + 256; }
        rv[t] = bv; ri[t] = bi;
        __syncthreads();
        for (int s = 128; s > 0; s >>= 1) {
            if (t < s) {
                if (rv[t + s] < rv[t] ||
                    (rv[t + s] == rv[t] && ri[t + s] < ri[t])) {
                    rv[t] = rv[t + s]; ri[t] = ri[t + s];
                }
            }
            __syncthreads();
        }
        if (t == 0) { idxout[(long)bn * KNN_ + kk] = ri[0]; d2[ri[0]] = 3.4e38f; }
        __syncthreads();
    }
}

// Edge features, written with padded row stride ldE (zero-filled pad columns):
// E[edge, 0:C] = nbr - ctr, E[edge, C:2C] = ctr, E[edge, 2C:ldE] = 0
__global__ void edge_build_kernel(const float* __restrict__ x,
                                  const int* __restrict__ idx,
                                  _Float16* __restrict__ E, int Cc, int ldE)
{
    const long t = (long)blockIdx.x * blockDim.x + threadIdx.x;
    if (t >= (long)BNK_ * ldE) return;
    const int  c2 = (int)(t % ldE);
    const long r  = t / ldE;             // edge row in [0, BNK)
    float val = 0.f;
    if (c2 < 2 * Cc) {
        const int  kk = (int)(r % KNN_);
        const long bn = r / KNN_;        // b*N + n
        const long b  = bn / N_;
        const int  j  = idx[bn * KNN_ + kk];
        if (c2 < Cc) val = x[(b * N_ + j) * (long)Cc + c2] - x[bn * Cc + c2];
        else         val = x[bn * Cc + (c2 - Cc)];
    }
    E[t] = (_Float16)val;
}

// Max over the K neighbors: [BN, K, C] -> [BN, C]
__global__ void maxpool_kernel(const float* __restrict__ pre,
                               float* __restrict__ outf,
                               _Float16* __restrict__ outh, int Cc)
{
    const long t = (long)blockIdx.x * blockDim.x + threadIdx.x;
    if (t >= (long)BN_ * Cc) return;
    const int  c  = (int)(t % Cc);
    const long bn = t / Cc;
    float m = -3.4e38f;
    for (int kk = 0; kk < KNN_; ++kk)
        m = fmaxf(m, pre[((long)bn * KNN_ + kk) * Cc + c]);
    if (outf) outf[t] = m;
    if (outh) outh[t] = (_Float16)m;
}

// rel_idx[b,i,j] = clip(int(|pos_i - pos_j| * 10), 0, 63)
__global__ void relidx_kernel(const float* __restrict__ pos, int* __restrict__ ridx)
{
    const long t = (long)blockIdx.x * blockDim.x + threadIdx.x;
    if (t >= (long)B_ * N_ * N_) return;
    const int  j  = (int)(t % N_);
    const long r  = t / N_;
    const int  i2 = (int)(r % N_);
    const long b  = r / N_;
    const float* pi = pos + (b * N_ + i2) * 3;
    const float* pj = pos + (b * N_ + j) * 3;
    const float dx = pi[0] - pj[0], dy = pi[1] - pj[1], dz = pi[2] - pj[2];
    int q = (int)(sqrtf(dx * dx + dy * dy + dz * dz) * 10.0f);
    q = q < 0 ? 0 : (q > 63 ? 63 : q);
    ridx[t] = q;
}

// Row softmax over N_, writes f16 probs into NPAD_-wide rows (pad cols = 0)
__global__ void softmax_kernel(const float* __restrict__ scores,
                               _Float16* __restrict__ probs)
{
    const int r = blockIdx.x, t = threadIdx.x;   // r in [0, B*H*N)
    const float* s = scores + (long)r * N_;
    _Float16* p = probs + (long)r * NPAD_;
    __shared__ float sh[256];
    __shared__ float mx, sum;

    float m = -3.4e38f;
    for (int j = t; j < N_; j += 256) m = fmaxf(m, s[j]);
    sh[t] = m; __syncthreads();
    for (int st = 128; st > 0; st >>= 1) {
        if (t < st) sh[t] = fmaxf(sh[t], sh[t + st]);
        __syncthreads();
    }
    if (t == 0) mx = sh[0];
    __syncthreads();

    float acc = 0.f;
    for (int j = t; j < N_; j += 256) acc += __expf(s[j] - mx);
    sh[t] = acc; __syncthreads();
    for (int st = 128; st > 0; st >>= 1) {
        if (t < st) sh[t] += sh[t + st];
        __syncthreads();
    }
    if (t == 0) sum = sh[0];
    __syncthreads();

    const float inv = 1.0f / sum;
    for (int j = t; j < NPAD_; j += 256)
        p[j] = (j < N_) ? (_Float16)(__expf(s[j] - mx) * inv) : (_Float16)0.0f;
}

// LayerNorm over D_=256 (block = one row, 256 threads), dual f32/f16 output
__global__ void ln_kernel(const float* __restrict__ x,
                          const float* __restrict__ g, const float* __restrict__ be,
                          float* __restrict__ outf, _Float16* __restrict__ outh)
{
    const int r = blockIdx.x, t = threadIdx.x;
    __shared__ float sh[256];
    __shared__ float mu, rsig;
    const float v = x[(long)r * D_ + t];
    sh[t] = v; __syncthreads();
    for (int s = 128; s > 0; s >>= 1) { if (t < s) sh[t] += sh[t + s]; __syncthreads(); }
    if (t == 0) mu = sh[0] * (1.0f / D_);
    __syncthreads();
    const float xm = v - mu;
    sh[t] = xm * xm; __syncthreads();
    for (int s = 128; s > 0; s >>= 1) { if (t < s) sh[t] += sh[t + s]; __syncthreads(); }
    if (t == 0) rsig = rsqrtf(sh[0] * (1.0f / D_) + 1e-5f);
    __syncthreads();
    const float y = xm * rsig * g[t] + be[t];
    outf[(long)r * D_ + t] = y;
    outh[(long)r * D_ + t] = (_Float16)y;
}

// Transposing / K-padding f32 -> f16 weight conversion into the Bt layout:
//   dst[p*dstStride + n*Kpad + k] = (k<K) ? f16(src[p*srcStride + k*sK + n*sN]) : 0
__global__ void cvt_bt_kernel(const float* __restrict__ src, long sK, long sN,
                              _Float16* __restrict__ dst,
                              int N, int K, int Kpad,
                              int batches, long srcStride, long dstStride)
{
    const long per = (long)N * Kpad;
    long t = (long)blockIdx.x * blockDim.x + threadIdx.x;
    if (t >= per * batches) return;
    const int  k = (int)(t % Kpad);
    const long r = t / Kpad;
    const long n = r % N;
    const long p = r / N;
    dst[p * dstStride + n * Kpad + k] =
        (k < K) ? (_Float16)src[p * srcStride + (long)k * sK + (long)n * sN]
                : (_Float16)0.0f;
}

// V transpose for the context GEMM: vt[(bh*DK + d)*NPAD + nk] (pad nk -> 0)
__global__ void vtrans_kernel(const _Float16* __restrict__ v16,
                              _Float16* __restrict__ vt)
{
    const long t = (long)blockIdx.x * blockDim.x + threadIdx.x;
    if (t >= (long)B_ * H_ * DK_ * NPAD_) return;
    const int  nk = (int)(t % NPAD_);
    const long r  = t / NPAD_;
    const int  d  = (int)(r % DK_);
    const long bh = r / DK_;
    const long b  = bh / H_;
    const int  h  = (int)(bh % H_);
    vt[t] = (nk < N_) ? v16[((long)b * N_ + nk) * D_ + h * DK_ + d]
                      : (_Float16)0.0f;
}

// ---------------------------------------------------------------------------
// Host-side helpers
// ---------------------------------------------------------------------------
static inline void launch_gemm(hipStream_t st,
    const _Float16* A, long saM, const _Float16* Bt, long sbN,
    float* Cf, _Float16* Ch, long ldc,
    const float* bias, const float* resid,
    int M, int N, int Kd, int mode,
    const float* bng = nullptr, const float* bnbe = nullptr,
    const float* bnm = nullptr, const float* bnv = nullptr)
{
    const long tiles = (long)((M + 31) / 32) * ((N + 31) / 32);
    const int blocks = (int)((tiles + 7) / 8);   // 8 waves (32x32 blocks) per CTA
    gemm_f16_kernel<<<blocks, 256, 0, st>>>(A, saM, Bt, sbN, Cf, Ch, ldc,
                                            bias, resid, M, N, Kd, mode,
                                            bng, bnbe, bnm, bnv);
}

static inline void cvt_bt(hipStream_t st, const float* src, long sK, long sN,
                          _Float16* dst, int N, int K, int Kpad,
                          int batches, long srcStride, long dstStride)
{
    const long elems = (long)N * Kpad * batches;
    cvt_bt_kernel<<<(int)((elems + 255) / 256), 256, 0, st>>>(
        src, sK, sN, dst, N, K, Kpad, batches, srcStride, dstStride);
}

// Input tensor order = depth-first dict order of setup_inputs()
enum {
    IN_POINTS = 0,
    EC1_W, EC1_B, EC1_G, EC1_BE, EC1_M, EC1_V,
    EC2_W, EC2_B, EC2_G, EC2_BE, EC2_M, EC2_V,
    INP_W, INP_B,
    LW_Q, LB_Q, LW_K, LB_K, LW_V, LB_V, LW_O, LB_O, L_REL,
    L_LN1G, L_LN1B, L_LN2G, L_LN2B, LW_1, LB_1, LW_2, LB_2,
    OUT_W1, OUT_B1, OUT_W2, OUT_B2
};

extern "C" void kernel_launch(void* const* d_in, const int* in_sizes, int n_in,
                              void* d_out, int out_size, void* d_ws, size_t ws_size,
                              hipStream_t stream)
{
    (void)in_sizes; (void)n_in; (void)out_size; (void)ws_size;

    const float* pts   = (const float*)d_in[IN_POINTS];
    const float* ec1W  = (const float*)d_in[EC1_W];
    const float* ec1b  = (const float*)d_in[EC1_B];
    const float* ec1g  = (const float*)d_in[EC1_G];
    const float* ec1be = (const float*)d_in[EC1_BE];
    const float* ec1m  = (const float*)d_in[EC1_M];
    const float* ec1v  = (const float*)d_in[EC1_V];
    const float* ec2W  = (const float*)d_in[EC2_W];
    const float* ec2b  = (const float*)d_in[EC2_B];
    const float* ec2g  = (const float*)d_in[EC2_G];
    const float* ec2be = (const float*)d_in[EC2_BE];
    const float* ec2m  = (const float*)d_in[EC2_M];
    const float* ec2v  = (const float*)d_in[EC2_V];
    const float* inpW  = (const float*)d_in[INP_W];
    const float* inpb  = (const float*)d_in[INP_B];
    const float* wq    = (const float*)d_in[LW_Q];
    const float* bq    = (const float*)d_in[LB_Q];
    const float* wk    = (const float*)d_in[LW_K];
    const float* bk    = (const float*)d_in[LB_K];
    const float* wv    = (const float*)d_in[LW_V];
    const float* bv    = (const float*)d_in[LB_V];
    const float* wo    = (const float*)d_in[LW_O];
    const float* bo    = (const float*)d_in[LB_O];
    const float* rel   = (const float*)d_in[L_REL];
    const float* ln1g  = (const float*)d_in[L_LN1G];
    const float* ln1b  = (const float*)d_in[L_LN1B];
    const float* ln2g  = (const float*)d_in[L_LN2G];
    const float* ln2b  = (const float*)d_in[L_LN2B];
    const float* w1    = (const float*)d_in[LW_1];
    const float* b1    = (const float*)d_in[LB_1];
    const float* w2    = (const float*)d_in[LW_2];
    const float* b2    = (const float*)d_in[LB_2];
    const float* oW1   = (const float*)d_in[OUT_W1];
    const float* ob1   = (const float*)d_in[OUT_B1];
    const float* oW2   = (const float*)d_in[OUT_W2];
    const float* ob2   = (const float*)d_in[OUT_B2];

    // ---- workspace bump allocator (~160 MB total) ----
    size_t off = 0;
    auto walloc = [&](size_t bytes) -> void* {
        off = (off + 255) & ~(size_t)255;
        void* p = (void*)((char*)d_ws + off);
        off += bytes;
        return p;
    };

    // Reused arenas:
    //   bigA: pre1 (20.5MB) -> E2 (20.5MB) -> scores (64MB)
    //   bigB: pre2 (41MB)   -> probs (32.8MB)
    void* bigA = walloc(64000000);
    void* bigB = walloc(40960000);
    float*    pre1   = (float*)bigA;
    _Float16* E2     = (_Float16*)bigA;
    float*    scores = (float*)bigA;
    float*    pre2   = (float*)bigB;
    _Float16* probs  = (_Float16*)bigB;

    int*       idx1   = (int*)walloc(sizeof(int) * BNK_);
    int*       idx2   = (int*)walloc(sizeof(int) * BNK_);
    _Float16*  E1     = (_Float16*)walloc(sizeof(_Float16) * (size_t)BNK_ * 32);
    float*     x1f    = (float*)walloc(sizeof(float) * (size_t)BN_ * 64);
    _Float16*  x2h    = (_Float16*)walloc(sizeof(_Float16) * (size_t)BN_ * 128);
    float*     featsf = (float*)walloc(sizeof(float) * (size_t)BN_ * D_);
    _Float16*  featsh = (_Float16*)walloc(sizeof(_Float16) * (size_t)BN_ * D_);
    int*       ridx   = (int*)walloc(sizeof(int) * (size_t)B_ * N_ * N_);
    _Float16*  q16    = (_Float16*)walloc(sizeof(_Float16) * (size_t)BN_ * D_);
    _Float16*  k16    = (_Float16*)walloc(sizeof(_Float16) * (size_t)BN_ * D_);
    _Float16*  v16    = (_Float16*)walloc(sizeof(_Float16) * (size_t)BN_ * D_);
    _Float16*  vt16   = (_Float16*)walloc(sizeof(_Float16) * (size_t)B_ * H_ * DK_ * NPAD_);
    _Float16*  ctx16  = (_Float16*)walloc(sizeof(_Float16) * (size_t)BN_ * D_);
    float*     tmpf   = (float*)walloc(sizeof(float) * (size_t)BN_ * D_);
    _Float16*  ffnh   = (_Float16*)walloc(sizeof(_Float16) * (size_t)BN_ * 4 * D_);
    _Float16*  headh  = (_Float16*)walloc(sizeof(_Float16) * (size_t)BN_ * 128);

    // f16 weights in Bt (n-major, K-contiguous, K padded to mult of 32) layout
    _Float16*  bt_ec1 = (_Float16*)walloc(sizeof(_Float16) * 64 * 32);
    _Float16*  bt_ec2 = (_Float16*)walloc(sizeof(_Float16) * 128 * 128);
    _Float16*  bt_inp = (_Float16*)walloc(sizeof(_Float16) * 256 * 128);
    _Float16*  bt_q   = (_Float16*)walloc(sizeof(_Float16) * (size_t)L_ * D_ * D_);
    _Float16*  bt_k   = (_Float16*)walloc(sizeof(_Float16) * (size_t)L_ * D_ * D_);
    _Float16*  bt_v   = (_Float16*)walloc(sizeof(_Float16) * (size_t)L_ * D_ * D_);
    _Float16*  bt_o   = (_Float16*)walloc(sizeof(_Float16) * (size_t)L_ * D_ * D_);
    _Float16*  bt_w1  = (_Float16*)walloc(sizeof(_Float16) * (size_t)L_ * 4 * D_ * D_);
    _Float16*  bt_w2  = (_Float16*)walloc(sizeof(_Float16) * (size_t)L_ * D_ * 4 * D_);
    _Float16*  bt_o1  = (_Float16*)walloc(sizeof(_Float16) * 128 * 256);
    _Float16*  bt_o2  = (_Float16*)walloc(sizeof(_Float16) * 3 * 128);

    // ---- weight conversion into Bt layouts ----
    // ec W[O,E]: B(k=e,n=o)=W[o*E+e] -> sK=1, sN=E
    cvt_bt(stream, ec1W, 1, 6,    bt_ec1, 64, 6, 32,    1, 0, 0);
    cvt_bt(stream, ec2W, 1, 128,  bt_ec2, 128, 128, 128, 1, 0, 0);
    // row-major [K,N] weights: sK=N, sN=1
    cvt_bt(stream, inpW, 256, 1,  bt_inp, 256, 128, 128, 1, 0, 0);
    cvt_bt(stream, wq, 256, 1,  bt_q,  256, 256, 256,  L_, (long)D_ * D_,     (long)D_ * D_);
    cvt_bt(stream, wk, 256, 1,  bt_k,  256, 256, 256,  L_, (long)D_ * D_,     (long)D_ * D_);
    cvt_bt(stream, wv, 256, 1,  bt_v,  256, 256, 256,  L_, (long)D_ * D_,     (long)D_ * D_);
    cvt_bt(stream, wo, 256, 1,  bt_o,  256, 256, 256,  L_, (long)D_ * D_,     (long)D_ * D_);
    cvt_bt(stream, w1, 1024, 1, bt_w1, 1024, 256, 256, L_, (long)D_ * 4 * D_, (long)D_ * 4 * D_);
    cvt_bt(stream, w2, 256, 1,  bt_w2, 256, 1024, 1024, L_, (long)D_ * 4 * D_, (long)D_ * 4 * D_);
    cvt_bt(stream, oW1, 128, 1, bt_o1, 128, 256, 256,  1, 0, 0);
    cvt_bt(stream, oW2, 3, 1,   bt_o2, 3, 128, 128,    1, 0, 0);

    // ---- EdgeConv 1 : pos[B,N,3] -> x1[B,N,64] ----
    knn_kernel<<<BN_, 256, 0, stream>>>(pts, idx1, 3);
    {
        const long elems = (long)BNK_ * 32;   // padded stride 32
        edge_build_kernel<<<(int)((elems + 255) / 256), 256, 0, stream>>>(pts, idx1, E1, 3, 32);
    }
    launch_gemm(stream, E1, 32, bt_ec1, 32, pre1, nullptr, 64,
                ec1b, nullptr, BNK_, 64, 32, EPI_BN_LEAKY, ec1g, ec1be, ec1m, ec1v);
    {
        const long elems = (long)BN_ * 64;
        maxpool_kernel<<<(int)((elems + 255) / 256), 256, 0, stream>>>(pre1, x1f, nullptr, 64);
    }

    // ---- EdgeConv 2 : x1[B,N,64] -> x2[B,N,128] ----
    knn_kernel<<<BN_, 256, 0, stream>>>(x1f, idx2, 64);
    {
        const long elems = (long)BNK_ * 128;
        edge_build_kernel<<<(int)((elems + 255) / 256), 256, 0, stream>>>(x1f, idx2, E2, 64, 128);
    }
    launch_gemm(stream, E2, 128, bt_ec2, 128, pre2, nullptr, 128,
                ec2b, nullptr, BNK_, 128, 128, EPI_BN_LEAKY, ec2g, ec2be, ec2m, ec2v);
    {
        const long elems = (long)BN_ * 128;
        maxpool_kernel<<<(int)((elems + 255) / 256), 256, 0, stream>>>(pre2, nullptr, x2h, 128);
    }

    // ---- input projection: feats = x2 @ Wp + bp ----
    launch_gemm(stream, x2h, 128, bt_inp, 128, featsf, featsh, D_,
                inpb, nullptr, BN_, D_, 128, EPI_BIAS);

    // ---- relative position indices (shared across layers) ----
    {
        const long elems = (long)B_ * N_ * N_;
        relidx_kernel<<<(int)((elems + 255) / 256), 256, 0, stream>>>(pts, ridx);
    }

    // ---- transformer layers ----
    const float scale = 1.0f / sqrtf((float)DK_);
    for (int i = 0; i < L_; ++i) {
        const long WQ = (long)i * D_ * D_;
        const long W1 = (long)i * D_ * 4 * D_;
        // Q, K, V projections (f16 out)
        launch_gemm(stream, featsh, D_, bt_q + WQ, D_, nullptr, q16, D_,
                    bq + i * D_, nullptr, BN_, D_, D_, EPI_BIAS);
        launch_gemm(stream, featsh, D_, bt_k + WQ, D_, nullptr, k16, D_,
                    bk + i * D_, nullptr, BN_, D_, D_, EPI_BIAS);
        launch_gemm(stream, featsh, D_, bt_v + WQ, D_, nullptr, v16, D_,
                    bv + i * D_, nullptr, BN_, D_, D_, EPI_BIAS);
        // scores + rel bias, softmax, V transpose, context
        attn_scores_kernel<<<dim3(32, B_ * H_), 256, 0, stream>>>(
            q16, k16, ridx, rel + (long)i * H_ * 64, scores, scale);
        softmax_kernel<<<B_ * H_ * N_, 256, 0, stream>>>(scores, probs);
        {
            const long elems = (long)B_ * H_ * DK_ * NPAD_;
            vtrans_kernel<<<(int)((elems + 255) / 256), 256, 0, stream>>>(v16, vt16);
        }
        attn_ctx_kernel<<<dim3(2, B_ * H_), 256, 0, stream>>>(probs, vt16, ctx16);
        // output projection + residual, then LN1
        launch_gemm(stream, ctx16, D_, bt_o + WQ, D_, tmpf, nullptr, D_,
                    bo + i * D_, featsf, BN_, D_, D_, EPI_BIAS);
        ln_kernel<<<BN_, 256, 0, stream>>>(tmpf, ln1g + i * D_, ln1b + i * D_,
                                           featsf, featsh);
        // FFN + residual, then LN2
        launch_gemm(stream, featsh, D_, bt_w1 + W1, D_, nullptr, ffnh, 4 * D_,
                    b1 + i * 4 * D_, nullptr, BN_, 4 * D_, D_, EPI_BIAS_RELU);
        launch_gemm(stream, ffnh, 4 * D_, bt_w2 + W1, 4 * D_, tmpf, nullptr, D_,
                    b2 + i * D_, featsf, BN_, D_, 4 * D_, EPI_BIAS);
        ln_kernel<<<BN_, 256, 0, stream>>>(tmpf, ln2g + i * D_, ln2b + i * D_,
                                           featsf, featsh);
    }

    // ---- output head: relu(feats @ W1 + b1) @ W2 + b2 + noisy_points ----
    launch_gemm(stream, featsh, D_, bt_o1, D_, nullptr, headh, 128,
                ob1, nullptr, BN_, 128, D_, EPI_BIAS_RELU);
    launch_gemm(stream, headh, 128, bt_o2, 128, (float*)d_out, nullptr, 3,
                ob2, pts, BN_, 3, 128, EPI_BIAS);
}
